// PhasorBlock_3315714752742
// MI455X (gfx1250) — matrix operand order repaired and checked by reading, not compiled, not run on hardware
//
#include <hip/hip_runtime.h>

#define PI_F 3.14159265358979323846f

typedef _Float16 h8   __attribute__((ext_vector_type(8)));
typedef _Float16 v16h __attribute__((ext_vector_type(16)));
typedef float    v8f  __attribute__((ext_vector_type(8)));
typedef float    f4   __attribute__((ext_vector_type(4)));
typedef unsigned int u32x4 __attribute__((ext_vector_type(4)));
typedef int          i32x4 __attribute__((ext_vector_type(4)));
typedef int          i32x8 __attribute__((ext_vector_type(8)));

static constexpr int Bb  = 4;
static constexpr int Ls  = 2048;
static constexpr int Ds  = 256;
static constexpr int Ps  = 32;
static constexpr int BL  = Bb * Ls;      // 8192
static constexpr int T2  = 64;           // mem2 chunk length
static constexpr int NC2 = Ls / T2;      // 32 chunks per batch
static constexpr int RS  = 128;          // scan chunk length
static constexpr int NCS = Ls / RS;      // 16 scan chunks per batch

// ---------------- TDM: 1-D contiguous global -> LDS copy --------------------
// D# per CDNA5 ISA §8.3/8.4: group0 = {flags, lds_addr, global_addr, type=2},
// group1 = {data_size=8B, tensor_dim0, tile_dim0, stride}. Arity differs per
// toolchain (ROCm7.2: 5 args, therock headers: 6 args) -> hedge on header.
#if defined(__has_include)
#if __has_include(<hip/amd_detail/amd_gfx1250_TDM.h>)
#define ATHENA_TDM_6ARG 1
#else
#define ATHENA_TDM_6ARG 0
#endif
#else
#define ATHENA_TDM_6ARG 0
#endif

__device__ __forceinline__ void tdm_copy_1d(unsigned ldsAddr, const void* gptr,
                                            unsigned nElem8B) {
  unsigned long long ga = (unsigned long long)(size_t)gptr;
  u32x4 g0;
  g0[0] = 1u;                                                  // count=1 valid user desc
  g0[1] = ldsAddr;                                             // LDS byte address
  g0[2] = (unsigned)(ga & 0xffffffffu);                        // global_addr lo
  g0[3] = (unsigned)((ga >> 32) & 0x01ffffffu) | (2u << 30);   // addr hi | type=2
  i32x8 g1;
  g1[0] = (3 << 16);                                           // data_size = 8 bytes
  g1[1] = (int)((nElem8B & 0xffffu) << 16);                    // tensor_dim0[15:0]
  g1[2] = (int)((nElem8B >> 16) & 0xffffu) | (1 << 16);        // dim0 hi | tensor_dim1=1
  g1[3] = (int)((nElem8B & 0xffffu) << 16);                    // tile_dim0
  g1[4] = 1;                                                   // tile_dim1=1, tile_dim2=0
  g1[5] = (int)nElem8B;                                        // tensor_dim0_stride lo
  g1[6] = 0;
  g1[7] = 0;
  i32x4 z4 = {0, 0, 0, 0};
#if ATHENA_TDM_6ARG
  i32x8 z8 = {0, 0, 0, 0, 0, 0, 0, 0};
  __builtin_amdgcn_tensor_load_to_lds(g0, g1, z4, z4, z8, 0);
#else
  __builtin_amdgcn_tensor_load_to_lds(g0, g1, z4, z4, 0);
#endif
}

// ---------- WMMA fragment helpers (layouts per ISA 7.12.2, wave32) ----------
__device__ __forceinline__ v16h ldA(const _Float16* p) {
  h8 lo = *(const h8*)p;
  h8 hi = *(const h8*)(p + 16);
  v16h r;
#pragma unroll
  for (int i = 0; i < 8; ++i) { r[i] = lo[i]; r[i + 8] = hi[i]; }
  return r;
}
__device__ __forceinline__ v16h ldB(const _Float16* p, int ld) {
  v16h r;
#pragma unroll
  for (int i = 0; i < 8; ++i) { r[i] = p[(size_t)i * ld]; r[i + 8] = p[(size_t)(16 + i) * ld]; }
  return r;
}
__device__ __forceinline__ v8f wmma16(v16h a, v16h b, v8f c) {
  return __builtin_amdgcn_wmma_f32_16x16x32_f16(false, a, false, b, (short)0, c, false, false);
}

// ---------- f32 -> f16 convert ----------
__global__ void cvt_f16(const float* __restrict__ src, _Float16* __restrict__ dst, int n) {
  int i = blockIdx.x * 256 + threadIdx.x;
  if (i < n) dst[i] = (_Float16)src[i];
}

// ---------- store gate: sigmoid(x @ w_gate + b), one wave per row ----------
__global__ __launch_bounds__(256) void gate_kernel(const float* __restrict__ x,
                                                   const float* __restrict__ wg,
                                                   const float* __restrict__ bg,
                                                   float* __restrict__ gate) {
  int wave = threadIdx.x >> 5, lane = threadIdx.x & 31;
  int m = blockIdx.x * 8 + wave;
  if (m >= BL) return;
  const float* row = x + (size_t)m * Ds;
  float s = 0.f;
#pragma unroll
  for (int j = 0; j < 8; ++j) s += row[lane * 8 + j] * wg[lane * 8 + j];
#pragma unroll
  for (int o = 16; o > 0; o >>= 1) s += __shfl_xor(s, o);
  if (lane == 0) gate[m] = 1.f / (1.f + expf(-(s + bg[0])));
}

// ---------- gate cumsum with clip>=1 : one wave per batch ----------
__global__ void gate_scan(const float* __restrict__ gate, float* __restrict__ gcum) {
  int wave = threadIdx.x >> 5, lane = threadIdx.x & 31;
  int b = wave;
  if (b >= Bb) return;
  float carry = 0.f;
  for (int it = 0; it < Ls / 32; ++it) {
    int l = it * 32 + lane;
    float v = gate[b * Ls + l];
#pragma unroll
    for (int d = 1; d < 32; d <<= 1) { float t = __shfl_up(v, d); if (lane >= d) v += t; }
    v += carry;
    gcum[b * Ls + l] = fmaxf(v, 1.0f);
    carry = __shfl(v, 31);
  }
}

// ---------- generic 3-pass chunked cumsum over L, C channels ----------
template <int C, int MODE>
__global__ void scan_p1(const float* __restrict__ in, float* __restrict__ part) {
  int u = blockIdx.x, b = u / NCS, c = u % NCS, ch = threadIdx.x;
  float s = 0.f;
  for (int r = 0; r < RS; ++r)
    s += in[((size_t)b * Ls + c * RS + r) * C + ch];
  part[(size_t)u * C + ch] = s;
}
template <int C>
__global__ void scan_p2(float* __restrict__ part) {
  int b = blockIdx.x, ch = threadIdx.x;
  float run = 0.f;
  for (int c = 0; c < NCS; ++c) {
    size_t id = ((size_t)(b * NCS + c)) * C + ch;
    float t = part[id]; part[id] = run; run += t;
  }
}
template <int C, int MODE>
__global__ void scan_p3(float* __restrict__ in, const float* __restrict__ part,
                        _Float16* __restrict__ outh) {
  int u = blockIdx.x, b = u / NCS, c = u % NCS, ch = threadIdx.x;
  float run = part[(size_t)u * C + ch];
  for (int r = 0; r < RS; ++r) {
    int l = c * RS + r;
    size_t id = ((size_t)b * Ls + l) * C + ch;
    run += in[id];
    if constexpr (MODE == 0) outh[id] = (_Float16)(run / (float)(l + 1));
    else                     in[id]  = run;
  }
}

// ---------- posoff[l,p] = pos_phases[l,:] @ w_off[D:,:] ----------
__global__ void posoff_kernel(const float* __restrict__ pos, const float* __restrict__ w_off,
                              float* __restrict__ posoff) {
  int i = blockIdx.x * 256 + threadIdx.x;
  if (i >= Ls * Ps) return;
  int l = i / Ps, p = i % Ps;
  float s = 0.f;
#pragma unroll
  for (int j = 0; j < Ps; ++j) s += pos[l * Ps + j] * w_off[(Ds + j) * Ps + p];
  posoff[i] = s;
}

// ---------- m1buf[m][0..31]=cos(ph)*v1, [32..63]=sin(ph)*v1 ----------
__global__ void m1build(const float* __restrict__ pos, const float* __restrict__ v1,
                        float* __restrict__ m1buf) {
  int i = blockIdx.x * 256 + threadIdx.x;
  if (i >= BL * Ps) return;
  int m = i / Ps, p = i % Ps, l = m % Ls;
  float ph = pos[l * Ps + p], v = v1[(size_t)m * Ps + p];
  m1buf[(size_t)m * 64 + p]      = cosf(ph) * v;
  m1buf[(size_t)m * 64 + 32 + p] = sinf(ph) * v;
}

// ---------- pos_ret (f16) = (m1c*qc + m1s*qs)/sqrt(P) ----------
__global__ void posret_kernel(const float* __restrict__ m1buf, const float* __restrict__ qbuf,
                              _Float16* __restrict__ pret) {
  int i = blockIdx.x * 256 + threadIdx.x;
  if (i >= BL * Ps) return;
  int m = i / Ps, p = i % Ps;
  float r = m1buf[(size_t)m * 64 + p] * qbuf[(size_t)m * 64 + p] +
            m1buf[(size_t)m * 64 + 32 + p] * qbuf[(size_t)m * 64 + 32 + p];
  pret[i] = (_Float16)(r * 0.17677669529663687f);
}

// ---------- generic WMMA GEMM: out[M,N] = A[M,K]@W[K,N] (+A2@W2) + epilogue ----------
template <int ACT, bool DUAL, int N, int K>
__global__ __launch_bounds__(256) void gemm_wmma(
    const _Float16* __restrict__ A,  const _Float16* __restrict__ W,
    const _Float16* __restrict__ A2, const _Float16* __restrict__ W2,
    const float* __restrict__ bias,
    float* __restrict__ outF, _Float16* __restrict__ outH, _Float16* __restrict__ outHT,
    const float* __restrict__ e0, const float* __restrict__ e1, int M) {
  int wave = threadIdx.x >> 5, lane = threadIdx.x & 31;
  constexpr int ntn = N >> 4;
  int tile = blockIdx.x * 8 + wave;
  if (tile >= (M >> 4) * ntn) return;
  int mt = tile / ntn, nt = tile % ntn;
  int m0 = mt << 4, n0 = nt << 4;
  int arow = m0 + (lane & 15);
  int koff = (lane >> 4) << 3;
  v8f acc = {};
#pragma unroll
  for (int k = 0; k < K; k += 32) {
    v16h a = ldA(A + (size_t)arow * K + k + koff);
    v16h b = ldB(W + (size_t)(k + koff) * N + n0 + (lane & 15), N);
    acc = wmma16(a, b, acc);
    if constexpr (DUAL) {
      v16h a2 = ldA(A2 + (size_t)arow * K + k + koff);
      v16h b2 = ldB(W2 + (size_t)(k + koff) * N + n0 + (lane & 15), N);
      acc = wmma16(a2, b2, acc);
    }
  }
  int grp = (lane >> 4) << 3;
  int nc = n0 + (lane & 15);
#pragma unroll
  for (int r = 0; r < 8; ++r) {
    int mr = m0 + r + grp;
    size_t idx = (size_t)mr * N + nc;
    float v = acc[r];
    if constexpr (ACT == 0) {               // plain f32 out + bias
      outF[idx] = v + bias[nc];
    } else if constexpr (ACT == 3) {        // phase pair -> f16 phi (and optional phiT)
      float t = tanhf(v + bias[nc]) * PI_F;
      float cs = cosf(t), sn = sinf(t);
      outH[(size_t)mr * 64 + nc]      = (_Float16)cs;
      outH[(size_t)mr * 64 + 32 + nc] = (_Float16)sn;
      if (outHT) {
        int bb = mr / Ls, l = mr % Ls;
        outHT[((size_t)bb * 64 + nc) * Ls + l]      = (_Float16)cs;
        outHT[((size_t)bb * 64 + 32 + nc) * Ls + l] = (_Float16)sn;
      }
    } else if constexpr (ACT == 4) {        // query: cos/sin(pos + tanh(..)*pi) -> f32
      int l = mr % Ls;
      float t = tanhf(v + e0[l * Ps + nc] + bias[nc]) * PI_F;
      float ph = e1[l * Ps + nc];
      outF[(size_t)mr * 64 + nc]      = cosf(ph + t);
      outF[(size_t)mr * 64 + 32 + nc] = sinf(ph + t);
    } else if constexpr (ACT == 5) {        // exact gelu -> f16
      float u = v + bias[nc];
      outH[idx] = (_Float16)(0.5f * u * (1.f + erff(u * 0.70710678118654752f)));
    } else if constexpr (ACT == 6) {        // gated values -> f16
      float u = (v + bias[nc]) * e0[mr];
      outH[idx] = (_Float16)u;
    } else if constexpr (ACT == 7) {        // final: + bias + residual x
      outF[idx] = v + bias[nc] + e0[idx];
    }
  }
}

// ---------- mem2 pass A: per-chunk local state S_loc[64,256] = phiK^T @ V ----------
// V chunk (32KB contiguous) staged to LDS by the Tensor Data Mover.
__global__ __launch_bounds__(256) void m2a_kernel(const _Float16* __restrict__ phiKT,
                                                  const _Float16* __restrict__ gvh,
                                                  float* __restrict__ Sbuf) {
  __shared__ __align__(16) _Float16 Vs[T2 * Ds];   // 32KB
  int u = blockIdx.x;              // b*NC2 + c
  int b = u / NC2, c = u % NC2, l0 = c * T2;
  int wave = threadIdx.x >> 5, lane = threadIdx.x & 31;
  if (threadIdx.x < 32)
    tdm_copy_1d((unsigned)(size_t)&Vs[0], gvh + ((size_t)b * Ls + l0) * Ds, T2 * Ds / 4);
  int mt = wave & 3;               // feature tile 0..3
  int njB = (wave >> 2) * 8;       // d tiles 0..7 or 8..15
  int koff = (lane >> 4) << 3;
  const _Float16* pA = phiKT + ((size_t)b * 64 + mt * 16 + (lane & 15)) * Ls + l0;
  v16h a0 = ldA(pA + koff);
  v16h a1 = ldA(pA + 32 + koff);
  if (threadIdx.x < 32) __builtin_amdgcn_s_wait_tensorcnt(0);
  __syncthreads();
#pragma unroll
  for (int j = 0; j < 8; ++j) {
    int n0 = (njB + j) * 16;
    v16h b0 = ldB(&Vs[(size_t)(koff)*Ds + n0 + (lane & 15)], Ds);
    v16h b1 = ldB(&Vs[(size_t)(32 + koff) * Ds + n0 + (lane & 15)], Ds);
    v8f s = {};
    s = wmma16(a0, b0, s);
    s = wmma16(a1, b1, s);
    int grp = (lane >> 4) << 3, nc = n0 + (lane & 15);
#pragma unroll
    for (int r = 0; r < 8; ++r)
      Sbuf[((size_t)u * 64 + mt * 16 + r + grp) * Ds + nc] = s[r];
  }
}

// ---------- mem2 pass B: exclusive prefix over chunks (in place) ----------
__global__ void m2b_kernel(float* __restrict__ Sbuf) {
  int i = blockIdx.x * 256 + threadIdx.x;      // over Bb*64*256
  if (i >= Bb * 64 * Ds) return;
  int b = i / (64 * Ds), rem = i % (64 * Ds), f = rem / Ds, d = rem % Ds;
  float run = 0.f;
  for (int c = 0; c < NC2; ++c) {
    size_t id = (((size_t)b * NC2 + c) * 64 + f) * Ds + d;
    float t = Sbuf[id]; Sbuf[id] = run; run += t;
  }
}

// ---------- mem2 pass C: out = mask(phiQ@phiK^T)@V + phiQ@Soff, scaled, += cbuf ----------
__global__ __launch_bounds__(256) void m2c_kernel(const _Float16* __restrict__ phiQ,
                                                  const _Float16* __restrict__ phiKT,
                                                  const _Float16* __restrict__ gvh,
                                                  const float* __restrict__ Sbuf,
                                                  const float* __restrict__ gcum,
                                                  float* __restrict__ cbuf) {
  __shared__ __align__(16) _Float16 attL[T2 * T2];  // 8KB
  __shared__ __align__(16) _Float16 Vs[T2 * Ds];    // 32KB (TDM)
  __shared__ __align__(16) _Float16 Ss[T2 * Ds];    // 32KB (converted f32 state)
  int u = blockIdx.x;
  int b = u / NC2, c = u % NC2, l0 = c * T2;
  int wave = threadIdx.x >> 5, lane = threadIdx.x & 31;
  int koff = (lane >> 4) << 3;
  int grp = (lane >> 4) << 3;

  if (threadIdx.x < 32)
    tdm_copy_1d((unsigned)(size_t)&Vs[0], gvh + ((size_t)b * Ls + l0) * Ds, T2 * Ds / 4);

  // stage 1: att[64,64] = phiQ_chunk @ phiK_chunk^T, causal-inclusive mask, -> LDS f16
#pragma unroll
  for (int t = 0; t < 2; ++t) {
    int at = wave * 2 + t, mtA = at >> 2, ntA = at & 3;
    const _Float16* pA = phiQ + ((size_t)b * Ls + l0 + mtA * 16 + (lane & 15)) * 64 + koff;
    v16h a0 = ldA(pA);
    v16h a1 = ldA(pA + 32);
    v16h b0 = ldB(phiKT + ((size_t)b * 64 + koff) * Ls + l0 + ntA * 16 + (lane & 15), Ls);
    v16h b1 = ldB(phiKT + ((size_t)b * 64 + 32 + koff) * Ls + l0 + ntA * 16 + (lane & 15), Ls);
    v8f s = {};
    s = wmma16(a0, b0, s);
    s = wmma16(a1, b1, s);
    int cc = ntA * 16 + (lane & 15);
#pragma unroll
    for (int r = 0; r < 8; ++r) {
      int rr = mtA * 16 + r + grp;
      attL[rr * T2 + cc] = (cc <= rr) ? (_Float16)s[r] : (_Float16)0.f;
    }
  }

  // cooperative coalesced load+convert of the f32 chunk state -> f16 LDS
  {
    const float* Sg = Sbuf + (size_t)u * 64 * Ds;
#pragma unroll
    for (int it = 0; it < 16; ++it) {
      int i = (it * 256 + threadIdx.x) * 4;
      f4 t = *(const f4*)(Sg + i);
#pragma unroll
      for (int e = 0; e < 4; ++e) Ss[i + e] = (_Float16)t[e];
    }
  }
  if (threadIdx.x < 32) __builtin_amdgcn_s_wait_tensorcnt(0);
  __syncthreads();

  // stage 2: out = att@V + phiQ@Soff  (all B fragments from LDS)
  int mt = wave & 3, njB = (wave >> 2) * 8;
  const _Float16* pL = &attL[(mt * 16 + (lane & 15)) * T2];
  v16h aa0 = ldA(pL + koff);
  v16h aa1 = ldA(pL + 32 + koff);
  const _Float16* pQ = phiQ + ((size_t)b * Ls + l0 + mt * 16 + (lane & 15)) * 64 + koff;
  v16h aq0 = ldA(pQ);
  v16h aq1 = ldA(pQ + 32);
#pragma unroll
  for (int j = 0; j < 8; ++j) {
    int n0 = (njB + j) * 16;
    v8f acc = {};
    v16h bv0 = ldB(&Vs[(size_t)(koff)*Ds + n0 + (lane & 15)], Ds);
    v16h bv1 = ldB(&Vs[(size_t)(32 + koff) * Ds + n0 + (lane & 15)], Ds);
    acc = wmma16(aa0, bv0, acc);
    acc = wmma16(aa1, bv1, acc);
    v16h bs0 = ldB(&Ss[(size_t)(koff)*Ds + n0 + (lane & 15)], Ds);
    v16h bs1 = ldB(&Ss[(size_t)(32 + koff) * Ds + n0 + (lane & 15)], Ds);
    acc = wmma16(aq0, bs0, acc);
    acc = wmma16(aq1, bs1, acc);
    int nc = n0 + (lane & 15);
#pragma unroll
    for (int r = 0; r < 8; ++r) {
      int l = l0 + mt * 16 + r + grp;
      float scale = rsqrtf(gcum[b * Ls + l]) * 0.17677669529663687f;  // 1/(sqrt(gc)*sqrt(P))
      cbuf[((size_t)b * Ls + l) * Ds + nc] += acc[r] * scale;
    }
  }
}

// ---------- LayerNorm over D=256, one wave per row, -> f16 ----------
__global__ __launch_bounds__(256) void ln_kernel(const float* __restrict__ cbuf,
                                                 const float* __restrict__ g,
                                                 const float* __restrict__ be,
                                                 _Float16* __restrict__ lnh) {
  int wave = threadIdx.x >> 5, lane = threadIdx.x & 31;
  int m = blockIdx.x * 8 + wave;
  if (m >= BL) return;
  const float* row = cbuf + (size_t)m * Ds;
  float v[8], s = 0.f, s2 = 0.f;
#pragma unroll
  for (int j = 0; j < 8; ++j) {
    v[j] = row[lane * 8 + j];
    s += v[j]; s2 += v[j] * v[j];
  }
#pragma unroll
  for (int o = 16; o > 0; o >>= 1) { s += __shfl_xor(s, o); s2 += __shfl_xor(s2, o); }
  float mu = s * (1.f / Ds);
  float var = s2 * (1.f / Ds) - mu * mu;
  float rstd = rsqrtf(var + 1e-5f);
#pragma unroll
  for (int j = 0; j < 8; ++j) {
    int d = lane * 8 + j;
    lnh[(size_t)m * Ds + d] = (_Float16)((v[j] - mu) * rstd * g[d] + be[d]);
  }
}

// =====================================================================
extern "C" void kernel_launch(void* const* d_in, const int* in_sizes, int n_in,
                              void* d_out, int out_size, void* d_ws, size_t ws_size,
                              hipStream_t stream) {
  (void)in_sizes; (void)n_in; (void)out_size; (void)ws_size;
  const float* x        = (const float*)d_in[0];
  const float* pos      = (const float*)d_in[1];
  const float* w_mem1v  = (const float*)d_in[2];
  const float* b_mem1v  = (const float*)d_in[3];
  const float* w_mem1o  = (const float*)d_in[4];
  const float* b_mem1o  = (const float*)d_in[5];
  const float* w_off    = (const float*)d_in[6];
  const float* b_off    = (const float*)d_in[7];
  const float* w_key    = (const float*)d_in[8];
  const float* b_key    = (const float*)d_in[9];
  const float* w_val    = (const float*)d_in[10];
  const float* b_val    = (const float*)d_in[11];
  const float* w_sk1    = (const float*)d_in[12];
  const float* b_sk1    = (const float*)d_in[13];
  const float* w_sk2    = (const float*)d_in[14];
  const float* b_sk2    = (const float*)d_in[15];
  const float* w_gate   = (const float*)d_in[16];
  const float* b_gate   = (const float*)d_in[17];
  const float* ln_g     = (const float*)d_in[18];
  const float* ln_b     = (const float*)d_in[19];
  const float* w_out    = (const float*)d_in[20];
  const float* b_out    = (const float*)d_in[21];
  float* out = (float*)d_out;

  char* wp = (char*)d_ws;
  auto carve = [&](size_t bytes) -> void* {
    void* p = (void*)wp;
    wp += (bytes + 255) & ~(size_t)255;
    return p;
  };
  _Float16* wv1h  = (_Float16*)carve((size_t)Ds * Ps * 2);
  _Float16* wkeyh = (_Float16*)carve((size_t)Ds * Ps * 2);
  _Float16* woffh = (_Float16*)carve((size_t)Ds * Ps * 2);
  _Float16* wsk2h = (_Float16*)carve((size_t)Ds * Ps * 2);
  _Float16* wm1oh = (_Float16*)carve((size_t)Ps * Ds * 2);
  _Float16* wvalh = (_Float16*)carve((size_t)Ds * Ds * 2);
  _Float16* wsk1h = (_Float16*)carve((size_t)2 * Ds * Ds * 2);
  _Float16* wouth = (_Float16*)carve((size_t)Ds * Ds * 2);
  _Float16* xh    = (_Float16*)carve((size_t)BL * Ds * 2);
  _Float16* ctxh  = (_Float16*)carve((size_t)BL * Ds * 2);
  _Float16* hh    = (_Float16*)carve((size_t)BL * Ds * 2);
  _Float16* gvh   = (_Float16*)carve((size_t)BL * Ds * 2);
  _Float16* lnh   = (_Float16*)carve((size_t)BL * Ds * 2);
  _Float16* phiQ  = (_Float16*)carve((size_t)BL * 64 * 2);
  _Float16* phiK  = (_Float16*)carve((size_t)BL * 64 * 2);
  _Float16* phiKT = (_Float16*)carve((size_t)BL * 64 * 2);
  _Float16* pret  = (_Float16*)carve((size_t)BL * Ps * 2);
  float* v1     = (float*)carve((size_t)BL * Ps * 4);
  float* qbuf   = (float*)carve((size_t)BL * 64 * 4);
  float* m1buf  = (float*)carve((size_t)BL * 64 * 4);
  float* posoff = (float*)carve((size_t)Ls * Ps * 4);
  float* gate   = (float*)carve((size_t)BL * 4);
  float* gcum   = (float*)carve((size_t)BL * 4);
  float* cbuf   = (float*)carve((size_t)BL * Ds * 4);
  float* Sbuf   = (float*)carve((size_t)Bb * NC2 * 64 * Ds * 4);
  float* partC  = (float*)carve((size_t)Bb * NCS * Ds * 4);
  float* partM  = (float*)carve((size_t)Bb * NCS * 64 * 4);

  auto cvt = [&](const float* s, _Float16* d, int n) {
    cvt_f16<<<(n + 255) / 256, 256, 0, stream>>>(s, d, n);
  };
  cvt(w_mem1v, wv1h, Ds * Ps);
  cvt(w_key,   wkeyh, Ds * Ps);
  cvt(w_off,   woffh, Ds * Ps);           // first D rows of w_off (x part)
  cvt(w_sk2,   wsk2h, Ds * Ps);
  cvt(w_mem1o, wm1oh, Ps * Ds);
  cvt(w_val,   wvalh, Ds * Ds);
  cvt(w_sk1,   wsk1h, 2 * Ds * Ds);
  cvt(w_out,   wouth, Ds * Ds);
  cvt(x,       xh,    BL * Ds);

  posoff_kernel<<<(Ls * Ps + 255) / 256, 256, 0, stream>>>(pos, w_off, posoff);
  gate_kernel<<<BL / 8, 256, 0, stream>>>(x, w_gate, b_gate, gate);
  gate_scan<<<1, 128, 0, stream>>>(gate, gcum);

  // ctx_avg scan (C=256) -> ctxh (f16)
  scan_p1<Ds, 0><<<Bb * NCS, Ds, 0, stream>>>((const float*)x, partC);
  scan_p2<Ds><<<Bb, Ds, 0, stream>>>(partC);
  scan_p3<Ds, 0><<<Bb * NCS, Ds, 0, stream>>>((float*)x, partC, ctxh);

  // projections from x (WMMA)
  gemm_wmma<0, false, Ps, Ds><<<(BL / 16) * (Ps / 16) / 8, 256, 0, stream>>>(
      xh, wv1h, nullptr, nullptr, b_mem1v, v1, nullptr, nullptr, nullptr, nullptr, BL);
  gemm_wmma<3, false, Ps, Ds><<<(BL / 16) * (Ps / 16) / 8, 256, 0, stream>>>(
      xh, wkeyh, nullptr, nullptr, b_key, nullptr, phiQ, nullptr, nullptr, nullptr, BL);
  gemm_wmma<6, false, Ds, Ds><<<(BL / 16) * (Ds / 16) / 8, 256, 0, stream>>>(
      xh, wvalh, nullptr, nullptr, b_val, nullptr, gvh, nullptr, gate, nullptr, BL);
  gemm_wmma<5, true, Ds, Ds><<<(BL / 16) * (Ds / 16) / 8, 256, 0, stream>>>(
      xh, wsk1h, ctxh, wsk1h + (size_t)Ds * Ds, b_sk1, nullptr, hh, nullptr, nullptr, nullptr,
      BL);
  gemm_wmma<3, false, Ps, Ds><<<(BL / 16) * (Ps / 16) / 8, 256, 0, stream>>>(
      hh, wsk2h, nullptr, nullptr, b_sk2, nullptr, phiK, phiKT, nullptr, nullptr, BL);

  // memory 1: build, scan (C=64 in place), query, retrieve, project
  m1build<<<(BL * Ps + 255) / 256, 256, 0, stream>>>(pos, v1, m1buf);
  scan_p1<64, 1><<<Bb * NCS, 64, 0, stream>>>(m1buf, partM);
  scan_p2<64><<<Bb, 64, 0, stream>>>(partM);
  scan_p3<64, 1><<<Bb * NCS, 64, 0, stream>>>(m1buf, partM, nullptr);
  gemm_wmma<4, false, Ps, Ds><<<(BL / 16) * (Ps / 16) / 8, 256, 0, stream>>>(
      xh, woffh, nullptr, nullptr, b_off, qbuf, nullptr, nullptr, posoff, pos, BL);
  posret_kernel<<<(BL * Ps + 255) / 256, 256, 0, stream>>>(m1buf, qbuf, pret);
  gemm_wmma<0, false, Ds, Ps><<<(BL / 16) * (Ds / 16) / 8, 256, 0, stream>>>(
      pret, wm1oh, nullptr, nullptr, b_mem1o, cbuf, nullptr, nullptr, nullptr, nullptr, BL);

  // memory 2: chunked linear attention (TDM-staged V chunks)
  m2a_kernel<<<Bb * NC2, 256, 0, stream>>>(phiKT, gvh, Sbuf);
  m2b_kernel<<<(Bb * 64 * Ds + 255) / 256, 256, 0, stream>>>(Sbuf);
  m2c_kernel<<<Bb * NC2, 256, 0, stream>>>(phiQ, phiKT, gvh, Sbuf, gcum, cbuf);

  // layernorm + output projection + residual
  ln_kernel<<<BL / 8, 256, 0, stream>>>(cbuf, ln_g, ln_b, lnh);
  gemm_wmma<7, false, Ds, Ds><<<(BL / 16) * (Ds / 16) / 8, 256, 0, stream>>>(
      lnh, wouth, nullptr, nullptr, b_out, out, nullptr, nullptr, x, nullptr, BL);
}
